// TokenLinear_49941879718509
// MI455X (gfx1250) — matrix operand order repaired and checked by reading, not compiled, not run on hardware
//
#include <hip/hip_runtime.h>
#include <hip/hip_bf16.h>

typedef __attribute__((ext_vector_type(16))) _Float16 v16h;
typedef __attribute__((ext_vector_type(8)))  _Float16 v8h;
typedef __attribute__((ext_vector_type(8)))  float    v8f;
typedef int v4i __attribute__((vector_size(16)));   // matches async-LDS builtin param

#define M_TOTAL   8192   // 4*2048 query rows
#define D_IN      1024
#define D_OUT     1024
#define T_TOK     1024
#define T2        2048   // 2*T_TOK combined tokens
#define RANK      32

#define GLOBAL_AS __attribute__((address_space(1)))
#define LDS_AS    __attribute__((address_space(3)))

#if defined(__AMDGCN__) && __has_builtin(__builtin_amdgcn_global_load_async_to_lds_b128)
#define USE_ASYNC_LDS 1
#else
#define USE_ASYNC_LDS 0
#endif

__device__ __forceinline__ void wait_async0() {
#if defined(__AMDGCN__)
#if __has_builtin(__builtin_amdgcn_s_wait_asynccnt)
    __builtin_amdgcn_s_wait_asynccnt(0);
#else
    asm volatile("s_wait_asynccnt 0x0" ::: "memory");
#endif
#endif
}

__device__ __forceinline__ float gelu_exact(float x) {
    return 0.5f * x * (1.0f + erff(x * 0.70710678118654752440f));
}

// ---------------------------------------------------------------------------
// f32 -> f16 bulk convert (8 elements / thread)
// ---------------------------------------------------------------------------
__global__ void cvt_f32_to_f16(const float* __restrict__ src,
                               _Float16* __restrict__ dst) {
    size_t i = ((size_t)blockIdx.x * blockDim.x + threadIdx.x) * 8;
    float4 a = *(const float4*)(src + i);
    float4 b = *(const float4*)(src + i + 4);
    v8h o;
    o[0] = (_Float16)a.x; o[1] = (_Float16)a.y;
    o[2] = (_Float16)a.z; o[3] = (_Float16)a.w;
    o[4] = (_Float16)b.x; o[5] = (_Float16)b.y;
    o[6] = (_Float16)b.z; o[7] = (_Float16)b.w;
    *(v8h*)(dst + i) = o;
}

// ---------------------------------------------------------------------------
// LoRA expand: one block per token row t.
//   out row t       = tokens[t]                  (f16 cast)
//   out row 1024+t  = gelu((tokens[t]@down)@up)  (f16 cast)
// transposed==1 writes out[col][tok] with pitch T2 (value matrix, pre-transposed
// so GEMM2's B operand columns are contiguous).
// ---------------------------------------------------------------------------
__global__ void lora_expand(const float* __restrict__ tokens,  // [T_TOK][D_IN]
                            const float* __restrict__ down,    // [D_IN][RANK]
                            const float* __restrict__ up,      // [RANK][D_IN]
                            _Float16* __restrict__ out,
                            int transposed) {
    __shared__ float row[D_IN];
    __shared__ float midp[8][RANK];
    __shared__ float mid[RANK];
    const int t   = blockIdx.x;
    const int tid = threadIdx.x;

    for (int k = tid; k < D_IN; k += 256) row[k] = tokens[(size_t)t * D_IN + k];
    __syncthreads();

    if (!transposed) {
        for (int k = tid; k < D_IN; k += 256)
            out[(size_t)t * D_IN + k] = (_Float16)row[k];
    } else {
        for (int k = tid; k < D_IN; k += 256)
            out[(size_t)k * T2 + t] = (_Float16)row[k];
    }

    const int j  = tid & 31;
    const int c  = tid >> 5;
    const int kb = c * (D_IN / 8);
    float p = 0.f;
    for (int k = 0; k < D_IN / 8; ++k)
        p += row[kb + k] * down[(size_t)(kb + k) * RANK + j];
    midp[c][j] = p;
    __syncthreads();
    if (tid < RANK) {
        float s = 0.f;
        #pragma unroll
        for (int cc = 0; cc < 8; ++cc) s += midp[cc][tid];
        mid[tid] = s;
    }
    __syncthreads();

    for (int k = tid; k < D_IN; k += 256) {
        float v = 0.f;
        #pragma unroll
        for (int r = 0; r < RANK; ++r) v += mid[r] * up[(size_t)r * D_IN + k];
        float g = gelu_exact(v);
        if (!transposed) out[(size_t)(T_TOK + t) * D_IN + k] = (_Float16)g;
        else             out[(size_t)k * T2 + (T_TOK + t)]   = (_Float16)g;
    }
}

// ---------------------------------------------------------------------------
// WMMA GEMM:  C[M][Ntot] = A[M][KDIM] * B[Ntot][KDIM]^T
// block = 256 threads (8 waves); block tile = 256(M) x 64(N); wave tile 32x64.
// B staged 64 cols x 64 k per buffer (8KB), double-buffered via async
// global->LDS (ASYNCcnt); 16 WMMAs per barrier, one barrier per 64-k step.
// k-loop kept at unroll 1 so accumulator tuples stay pinned (no rotation
// copies -> no WMMA->VALU hazard NOPs in the steady state).
// ---------------------------------------------------------------------------
template <int KDIM, bool STORE_F16, bool SUMSQ>
__global__ void __launch_bounds__(256) wmma_gemm(const _Float16* __restrict__ A,
                                                 const _Float16* __restrict__ B,
                                                 void* __restrict__ Cout,
                                                 float* __restrict__ rss,
                                                 int Ntot) {
    // two buffers of 64 rows x 64 k f16 (row pitch 64 elem = 128B)
    __shared__ __attribute__((aligned(32))) _Float16 Bs[2 * 64 * 64];

    const int lane = threadIdx.x & 31;
    const int wave = threadIdx.x >> 5;
    const int half = lane >> 4;   // 0/1 lane group
    const int ln   = lane & 15;   // 0..15
    const int m0   = blockIdx.x * 256 + wave * 32;
    const int n0   = blockIdx.y * 64;

    const _Float16* arow0 = A + (size_t)(m0 + ln) * KDIM + 16 * half;
    const _Float16* arow1 = arow0 + (size_t)16 * KDIM;

    // staging: 64 rows x 8 chunks(8 f16) = 512 chunks; thread does chunk tid, tid+256
    const int c0n = threadIdx.x >> 3;          // row of chunk tid
    const int c0k = (threadIdx.x & 7) * 8;     // k offset of chunk tid
    const _Float16* gs0 = B + (size_t)(n0 + c0n) * KDIM + c0k;
    const _Float16* gs1 = B + (size_t)(n0 + 32 + c0n) * KDIM + c0k;  // chunk tid+256
    _Float16* ls0 = &Bs[c0n * 64 + c0k];
    _Float16* ls1 = &Bs[(32 + c0n) * 64 + c0k];

    v8f acc0[4], acc1[4];
    #pragma unroll
    for (int j = 0; j < 4; ++j) { acc0[j] = (v8f){}; acc1[j] = (v8f){}; }

    constexpr int STEPS = KDIM / 64;

    auto stage = [&](int i) {  // async-load 64x64 tile i into buffer i&1
        const int kb  = i * 64;
        const int bo  = (i & 1) * 64 * 64;
#if USE_ASYNC_LDS
        __builtin_amdgcn_global_load_async_to_lds_b128(
            (GLOBAL_AS v4i*)(gs0 + kb), (LDS_AS v4i*)(ls0 + bo), 0, 0);
        __builtin_amdgcn_global_load_async_to_lds_b128(
            (GLOBAL_AS v4i*)(gs1 + kb), (LDS_AS v4i*)(ls1 + bo), 0, 0);
#else
        v8h t0 = *(const v8h*)(gs0 + kb);
        v8h t1 = *(const v8h*)(gs1 + kb);
        *(v8h*)(ls0 + bo) = t0;
        *(v8h*)(ls1 + bo) = t1;
#endif
    };

    auto compute32 = [&](int kb, int bufbase, int ksub) {
        v8h alo0 = *(const v8h*)(arow0 + kb + ksub);
        v8h ahi0 = *(const v8h*)(arow0 + kb + ksub + 8);
        v8h alo1 = *(const v8h*)(arow1 + kb + ksub);
        v8h ahi1 = *(const v8h*)(arow1 + kb + ksub + 8);
        v16h a0, a1;
        #pragma unroll
        for (int i = 0; i < 8; ++i) {
            a0[i] = alo0[i]; a0[8 + i] = ahi0[i];
            a1[i] = alo1[i]; a1[8 + i] = ahi1[i];
        }
        v16h b[4];
        #pragma unroll
        for (int j = 0; j < 4; ++j)
            b[j] = *(const v16h*)(&Bs[bufbase + (j * 16 + ln) * 64 + ksub + 16 * half]);
        #pragma unroll
        for (int j = 0; j < 4; ++j) {
            acc0[j] = __builtin_amdgcn_wmma_f32_16x16x32_f16(
                false, a0, false, b[j], (short)0, acc0[j], false, false);
            acc1[j] = __builtin_amdgcn_wmma_f32_16x16x32_f16(
                false, a1, false, b[j], (short)0, acc1[j], false, false);
        }
    };

    stage(0);
    wait_async0();
    __syncthreads();
    #pragma unroll 1
    for (int i = 0; i < STEPS - 1; ++i) {
        stage(i + 1);
        const int bo = (i & 1) * 64 * 64;
        compute32(i * 64, bo, 0);
        compute32(i * 64, bo, 32);
        wait_async0();
        __syncthreads();
    }
    {
        const int i = STEPS - 1;
        const int bo = (i & 1) * 64 * 64;
        compute32(i * 64, bo, 0);
        compute32(i * 64, bo, 32);
    }

    // epilogue: C layout -> row (half*8 + r), col (j*16 + ln)
    auto epilogue = [&](v8f (&acc)[4], int mrow) {
        #pragma unroll
        for (int r = 0; r < 8; ++r) {
            float ss = 0.f;
            #pragma unroll
            for (int j = 0; j < 4; ++j) {
                float v = acc[j][r];
                size_t idx = (size_t)(mrow + r) * Ntot + n0 + j * 16 + ln;
                if (STORE_F16) ((_Float16*)Cout)[idx] = (_Float16)v;
                else           ((float*)Cout)[idx]    = v;
                ss += v * v;
            }
            if (SUMSQ) {
                #pragma unroll
                for (int off = 1; off < 16; off <<= 1)
                    ss += __shfl_xor(ss, off, 32);
                if (ln == 0) atomicAdd(&rss[mrow + r], ss);
            }
        }
    };
    epilogue(acc0, m0 + half * 8);
    epilogue(acc1, m0 + 16 + half * 8);
}

// ---------------------------------------------------------------------------
// In-place row-normalized exact GELU on S (f16), 8 elems / thread.
// ---------------------------------------------------------------------------
__global__ void gelu_norm(_Float16* __restrict__ S,
                          const float* __restrict__ rss) {
    size_t i = ((size_t)blockIdx.x * blockDim.x + threadIdx.x) * 8;
    int m = (int)(i >> 11);  // row length T2 = 2048
    float scale = 45.254833995939045f * rsqrtf(fmaxf(rss[m], 1e-30f));
    v8h v = *(const v8h*)(S + i);
    v8h o;
    #pragma unroll
    for (int k = 0; k < 8; ++k) {
        float x = (float)v[k] * scale;
        o[k] = (_Float16)gelu_exact(x);
    }
    *(v8h*)(S + i) = o;
}

// ---------------------------------------------------------------------------
extern "C" void kernel_launch(void* const* d_in, const int* in_sizes, int n_in,
                              void* d_out, int out_size, void* d_ws, size_t ws_size,
                              hipStream_t stream) {
    const float* x            = (const float*)d_in[0];
    const float* key_tokens   = (const float*)d_in[1];
    const float* value_tokens = (const float*)d_in[2];
    const float* key_down     = (const float*)d_in[3];
    const float* key_up       = (const float*)d_in[4];
    const float* value_down   = (const float*)d_in[5];
    const float* value_up     = (const float*)d_in[6];
    float* out = (float*)d_out;

    char* ws = (char*)d_ws;
    _Float16* Xh  = (_Float16*)ws;                 // [8192][1024] f16  16 MB
    ws += (size_t)M_TOTAL * D_IN * 2;
    _Float16* CK  = (_Float16*)ws;                 // [2048][1024] f16   4 MB
    ws += (size_t)T2 * D_IN * 2;
    _Float16* CVT = (_Float16*)ws;                 // [1024][2048] f16   4 MB
    ws += (size_t)D_OUT * T2 * 2;
    _Float16* S   = (_Float16*)ws;                 // [8192][2048] f16  32 MB
    ws += (size_t)M_TOTAL * T2 * 2;
    float* rss    = (float*)ws;                    // [8192] f32        32 KB

    (void)hipMemsetAsync(rss, 0, M_TOTAL * sizeof(float), stream);

    cvt_f32_to_f16<<<(M_TOTAL * D_IN) / (256 * 8), 256, 0, stream>>>(x, Xh);

    lora_expand<<<T_TOK, 256, 0, stream>>>(key_tokens, key_down, key_up, CK, 0);
    lora_expand<<<T_TOK, 256, 0, stream>>>(value_tokens, value_down, value_up, CVT, 1);

    // S = Xh @ CK^T  (+ row sum of squares)
    wmma_gemm<D_IN, true, true>
        <<<dim3(M_TOTAL / 256, T2 / 64), 256, 0, stream>>>(Xh, CK, S, rss, T2);

    // S = gelu(S * sqrt(2048)/||row||)
    gelu_norm<<<(M_TOTAL * T2) / (256 * 8), 256, 0, stream>>>(S, rss);

    // out = S @ CVT^T  (f32 output)
    wmma_gemm<T2, false, false>
        <<<dim3(M_TOTAL / 256, D_OUT / 64), 256, 0, stream>>>(S, CVT, out, nullptr, D_OUT);
}
